// SelfAttention_53901839565216
// MI455X (gfx1250) — matrix operand order repaired and checked by reading, not compiled, not run on hardware
//
#include <hip/hip_runtime.h>
#include <hip/hip_bf16.h>
#include <stdint.h>

// ---------------- problem constants (match reference) ----------------
#define BB 4
#define TT 2048
#define CC 512
#define HH 8
#define HD 64
#define MM (BB * TT)   // 8192 rows

typedef __attribute__((ext_vector_type(16))) __bf16 v16bf;
typedef __attribute__((ext_vector_type(4)))  __bf16 bf16x4;
typedef __attribute__((ext_vector_type(2)))  __bf16 bf16x2;
typedef __attribute__((ext_vector_type(8)))  float  v8f;

__device__ __forceinline__ int lane_id() { return threadIdx.x & 31; }

__device__ __forceinline__ v8f wmma_bf16(v16bf a, v16bf b, v8f c) {
    // D = A(16x32 bf16) * B(32x16 bf16) + C(16x16 f32)
    return __builtin_amdgcn_wmma_f32_16x16x32_bf16(
        /*neg_a=*/false, a, /*neg_b=*/false, b,
        /*c_mod=*/(short)0, c, /*reuse_a=*/false, /*reuse_b=*/false);
}

// 32-bit LDS byte address for async/inline-asm use (generic LDS ptr truncates
// to the LDS offset per the flat-aperture rules).
__device__ __forceinline__ uint32_t lds_addr32(const void* p) {
    return (uint32_t)(uintptr_t)p;
}

// CDNA5 async global->LDS copy, 16 bytes per lane; tracked by ASYNCcnt.
__device__ __forceinline__ void async_copy_b128(uint32_t lds_off, const void* gptr) {
    uint64_t ga = (uint64_t)(uintptr_t)gptr;
    asm volatile("global_load_async_to_lds_b128 %0, %1, off"
                 :: "v"(lds_off), "v"(ga) : "memory");
}

// ---- A fragment (16x32, bf16) from row-major bf16 src [.. , lda] ----
// ISA 16-bit A layout: m = l%16 ; k = (v%4)*2 + (v/4)*16 + (l/16)*8 + h
__device__ __forceinline__ v16bf load_a_bf16(const __bf16* __restrict__ p,
                                             int lda, int m0, int k0) {
    int l = lane_id();
    const __bf16* row = p + (size_t)(m0 + (l & 15)) * lda;
    int khalf = (l >> 4) * 8;
    v16bf a;
#pragma unroll
    for (int v = 0; v < 8; ++v) {
        int k = k0 + ((v & 3) * 2) + ((v >> 2) * 16) + khalf;
        bf16x2 t = *(const bf16x2*)(row + k);
        a[2 * v] = t[0];
        a[2 * v + 1] = t[1];
    }
    return a;
}

// ---- B fragment (32x16) where B[k][n] = src[n][k] (src row-major [*, ld]) ----
// ISA 16-bit B layout: n = l%16 ; k = k0 + (v%4)*2 + (v/4)*16 + (l/16)*8 + h  (pairs contiguous)
__device__ __forceinline__ v16bf load_b_nk_bf16(const __bf16* __restrict__ p,
                                                int ld, int n0, int k0) {
    int l = lane_id();
    const __bf16* row = p + (size_t)(n0 + (l & 15)) * ld;
    int kb = k0 + (l >> 4) * 8;
    v16bf b;
#pragma unroll
    for (int v = 0; v < 8; ++v) {
        bf16x2 t = *(const bf16x2*)(row + kb + ((v & 3) * 2) + ((v >> 2) * 16));
        b[2 * v] = t[0];
        b[2 * v + 1] = t[1];
    }
    return b;
}

// =====================================================================
// Kernel 0: elementwise f32 -> bf16 (x4 per thread); n % 4 == 0
// =====================================================================
__global__ void cvt_bf16_kernel(const float* __restrict__ src,
                                __bf16* __restrict__ dst, int n) {
    int i = (blockIdx.x * blockDim.x + threadIdx.x) * 4;
    if (i >= n) return;
    float4 t = *(const float4*)(src + i);
    bf16x4 r;
    r[0] = (__bf16)t.x; r[1] = (__bf16)t.y;
    r[2] = (__bf16)t.z; r[3] = (__bf16)t.w;
    *(bf16x4*)(dst + i) = r;
}

// =====================================================================
// Kernel 1: Y = X @ W^T + bias (bf16 in, bf16 out)
//   transposed==0 : head-major [B,H,T,HD]   (q, k)
//   transposed==1 : [B,H,HD,T]              (v -> fed as contiguous B later)
// 32x64 output tile per wave (B fragments reused across 2 row-tiles).
// grid: 256 blocks * 8 waves = 2048 wave-tiles = (M/32)*(C/64)
// =====================================================================
__global__ void proj_kernel(const __bf16* __restrict__ X,
                            const __bf16* __restrict__ W,
                            const float* __restrict__ bias,
                            __bf16* __restrict__ Y,
                            int transposed) {
    int wid = blockIdx.x * (blockDim.x >> 5) + (threadIdx.x >> 5);
    int n0 = (wid & 7) * 64;         // column tile == one head (HD==64)
    int m0 = (wid >> 3) * 32;        // row tile (two 16-row fragments)

    v8f c00 = {}, c01 = {}, c02 = {}, c03 = {};
    v8f c10 = {}, c11 = {}, c12 = {}, c13 = {};
#pragma unroll 4
    for (int k0 = 0; k0 < CC; k0 += 32) {
        if (k0 + 32 < CC) {
            __builtin_prefetch(X + (size_t)(m0 + (lane_id() & 15)) * CC + k0 + 32, 0, 3);
            __builtin_prefetch(X + (size_t)(m0 + 16 + (lane_id() & 15)) * CC + k0 + 32, 0, 3);
        }
        // load all fragments first -> one clause + back-to-back WMMAs
        v16bf a0 = load_a_bf16(X, CC, m0,      k0);
        v16bf a1 = load_a_bf16(X, CC, m0 + 16, k0);
        v16bf b0 = load_b_nk_bf16(W, CC, n0 +  0, k0);
        v16bf b1 = load_b_nk_bf16(W, CC, n0 + 16, k0);
        v16bf b2 = load_b_nk_bf16(W, CC, n0 + 32, k0);
        v16bf b3 = load_b_nk_bf16(W, CC, n0 + 48, k0);
        c00 = wmma_bf16(a0, b0, c00);
        c10 = wmma_bf16(a1, b0, c10);
        c01 = wmma_bf16(a0, b1, c01);
        c11 = wmma_bf16(a1, b1, c11);
        c02 = wmma_bf16(a0, b2, c02);
        c12 = wmma_bf16(a1, b2, c12);
        c03 = wmma_bf16(a0, b3, c03);
        c13 = wmma_bf16(a1, b3, c13);
    }

    int l = lane_id();
    int nl = l & 15;
    int moff = (l >> 4) * 8;
    int h = n0 >> 6;
    v8f acc[2][4] = {{c00, c01, c02, c03}, {c10, c11, c12, c13}};
#pragma unroll
    for (int r = 0; r < 2; ++r) {
#pragma unroll
        for (int j = 0; j < 4; ++j) {
            int n = n0 + j * 16 + nl;
            float bv = bias[n];
            int hd = n & 63;
#pragma unroll
            for (int v = 0; v < 8; ++v) {
                int m = m0 + r * 16 + v + moff;
                int b = m >> 11;             // T = 2048
                int t = m & (TT - 1);
                __bf16 val = (__bf16)(acc[r][j][v] + bv);
                if (transposed)
                    Y[((size_t)(b * HH + h) * HD + hd) * TT + t] = val;
                else
                    Y[((size_t)(b * HH + h) * TT + t) * HD + hd] = val;
            }
        }
    }
}

// =====================================================================
// Kernel 2: flash attention for one branch.
// q,k: bf16 [B,H,T,HD]; vt: bf16 [B,H,HD,T]; out: bf16 [B,T,C].
// 8 waves/block, 16 queries/wave -> 128 queries/block; grid = B*H*T/128.
// K/V^T tiles double-buffered in LDS via async global->LDS copies.
// =====================================================================
__global__ void attn_kernel(const __bf16* __restrict__ q,
                            const __bf16* __restrict__ k,
                            const __bf16* __restrict__ vt,
                            __bf16* __restrict__ out) {
    __shared__ __bf16 kbuf[2][32 * HD];    // 32 keys x 64 hd      (4 KB each)
    __shared__ __bf16 vbuf[2][HD * 32];    // 64 hd  x 32 keys     (4 KB each)
    __shared__ float  pstage[8][16 * 32];  // per-wave P staging   (16 KB)

    int tid = threadIdx.x;
    int w = tid >> 5;
    int l = lane_id();
    int bh = blockIdx.x >> 4;              // T/128 = 16 query chunks
    int qchunk = blockIdx.x & 15;
    int q0 = qchunk * 128 + w * 16;        // query row within this head

    const __bf16* qh = q  + (size_t)bh * TT * HD;
    const __bf16* kh = k  + (size_t)bh * TT * HD;
    const __bf16* vh = vt + (size_t)bh * HD * TT;

    // Q fragments (A-layout), reused for all key blocks
    v16bf aq0 = load_a_bf16(qh, HD, q0, 0);
    v16bf aq1 = load_a_bf16(qh, HD, q0, 32);

    v8f o0 = {}, o1 = {}, o2 = {}, o3 = {};
    float Mx[8], Ls[8];
#pragma unroll
    for (int i = 0; i < 8; ++i) { Mx[i] = -1e30f; Ls[i] = 0.0f; }

    float* pst = &pstage[w][0];
    const float scale = 0.125f;            // 1/sqrt(64)
    int nl = l & 15;
    int moff = (l >> 4) * 8;

    // staging assignment: 256 threads x 16B = one 4KB tile per array
    int vrow = tid >> 2;                   // hd row 0..63
    int vchk = tid & 3;                    // 8-elem chunk within 32 keys
    uint32_t klds0 = lds_addr32(&kbuf[0][0]) + tid * 16;
    uint32_t klds1 = lds_addr32(&kbuf[1][0]) + tid * 16;
    uint32_t vlds0 = lds_addr32(&vbuf[0][0]) + (vrow * 32 + vchk * 8) * 2;
    uint32_t vlds1 = lds_addr32(&vbuf[1][0]) + (vrow * 32 + vchk * 8) * 2;

    // prologue: stage key-block 0 into buffer 0
    async_copy_b128(klds0, kh + tid * 8);
    async_copy_b128(vlds0, vh + (size_t)vrow * TT + vchk * 8);

    const int NKB = TT / 32;               // 64 key blocks
    for (int kb = 0; kb < NKB; ++kb) {
        int cur = kb & 1;
        // issue next block's tiles into the other buffer (overlap with compute)
        if (kb + 1 < NKB) {
            int nb0 = (kb + 1) * 32;
            async_copy_b128(cur ? klds0 : klds1, kh + (size_t)nb0 * HD + tid * 8);
            async_copy_b128(cur ? vlds0 : vlds1,
                            vh + (size_t)vrow * TT + nb0 + vchk * 8);
            asm volatile("s_wait_asynccnt 0x2" ::: "memory");
        } else {
            asm volatile("s_wait_asynccnt 0x0" ::: "memory");
        }
        __syncthreads();                   // current tiles visible to all waves

        const __bf16* kt = &kbuf[cur][0];  // [32][64]
        const __bf16* vtt = &vbuf[cur][0]; // [64][32]

        // ---- S = Q * K^T for 32 keys (two 16-wide n-tiles) ----
        v16bf bk00 = load_b_nk_bf16(kt, HD, 0,  0);
        v16bf bk01 = load_b_nk_bf16(kt, HD, 0,  32);
        v16bf bk10 = load_b_nk_bf16(kt, HD, 16, 0);
        v16bf bk11 = load_b_nk_bf16(kt, HD, 16, 32);
        v8f s0 = {}, s1 = {};
        s0 = wmma_bf16(aq0, bk00, s0);
        s1 = wmma_bf16(aq0, bk10, s1);
        s0 = wmma_bf16(aq1, bk01, s0);
        s1 = wmma_bf16(aq1, bk11, s1);

        // ---- online softmax (rows live across 16-lane half-wave) ----
        float alpha[8];
#pragma unroll
        for (int vv = 0; vv < 8; ++vv) {
            float e0 = s0[vv] * scale;
            float e1 = s1[vv] * scale;
            float mx = fmaxf(e0, e1);
#pragma unroll
            for (int off = 1; off < 16; off <<= 1)
                mx = fmaxf(mx, __shfl_xor(mx, off, 32));
            float mn = fmaxf(Mx[vv], mx);
            float p0 = __expf(e0 - mn);
            float p1 = __expf(e1 - mn);
            float al = __expf(Mx[vv] - mn);
            float rs = p0 + p1;
#pragma unroll
            for (int off = 1; off < 16; off <<= 1)
                rs += __shfl_xor(rs, off, 32);
            Ls[vv] = Ls[vv] * al + rs;
            Mx[vv] = mn;
            alpha[vv] = al;
            // stage P (S-layout -> LDS row-major 16x32)
            pst[(vv + moff) * 32 + nl]      = p0;
            pst[(vv + moff) * 32 + 16 + nl] = p1;
        }

#pragma unroll
        for (int vv = 0; vv < 8; ++vv) {
            o0[vv] *= alpha[vv]; o1[vv] *= alpha[vv];
            o2[vv] *= alpha[vv]; o3[vv] *= alpha[vv];
        }

        asm volatile("s_wait_dscnt 0" ::: "memory");

        // ---- read P back in A-fragment layout (bf16) ----
        v16bf ap;
#pragma unroll
        for (int vv = 0; vv < 8; ++vv) {
            int kk = ((vv & 3) * 2) + ((vv >> 2) * 16) + ((l >> 4) * 8);
            ap[2 * vv]     = (__bf16)pst[(l & 15) * 32 + kk];
            ap[2 * vv + 1] = (__bf16)pst[(l & 15) * 32 + kk + 1];
        }

        // ---- O += P * V ; B[k][n] = V^T[n][k], contiguous in LDS ----
        v16bf bv0 = load_b_nk_bf16(vtt, 32, 0,  0);
        v16bf bv1 = load_b_nk_bf16(vtt, 32, 16, 0);
        v16bf bv2 = load_b_nk_bf16(vtt, 32, 32, 0);
        v16bf bv3 = load_b_nk_bf16(vtt, 32, 48, 0);
        o0 = wmma_bf16(ap, bv0, o0);
        o1 = wmma_bf16(ap, bv1, o1);
        o2 = wmma_bf16(ap, bv2, o2);
        o3 = wmma_bf16(ap, bv3, o3);

        __syncthreads();                   // all waves done reading buffers
    }

    // ---- normalize and store bf16 into [B,T,C] ----
    int b = bh >> 3;           // H = 8
    int h = bh & 7;
    float linv[8];
#pragma unroll
    for (int vv = 0; vv < 8; ++vv) linv[vv] = 1.0f / Ls[vv];

    v8f oo[4] = {o0, o1, o2, o3};
#pragma unroll
    for (int j = 0; j < 4; ++j) {
        int hd = j * 16 + nl;
#pragma unroll
        for (int vv = 0; vv < 8; ++vv) {
            int t = q0 + vv + moff;
            out[((size_t)b * TT + t) * CC + h * 64 + hd] = (__bf16)(oo[j][vv] * linv[vv]);
        }
    }
}

// =====================================================================
// Kernel 3: out = A1 @ W1^T + b1 + A2 @ W2^T + b2   (f32 output [M,C])
// A1,A2,W1,W2: bf16; 32x64 tile per wave; 256 blocks * 8 waves.
// =====================================================================
__global__ void out_proj_kernel(const __bf16* __restrict__ A1,
                                const __bf16* __restrict__ W1,
                                const float* __restrict__ b1,
                                const __bf16* __restrict__ A2,
                                const __bf16* __restrict__ W2,
                                const float* __restrict__ b2,
                                float* __restrict__ out) {
    int wid = blockIdx.x * (blockDim.x >> 5) + (threadIdx.x >> 5);
    int n0 = (wid & 7) * 64;
    int m0 = (wid >> 3) * 32;

    v8f c00 = {}, c01 = {}, c02 = {}, c03 = {};
    v8f c10 = {}, c11 = {}, c12 = {}, c13 = {};
#pragma unroll
    for (int pass = 0; pass < 2; ++pass) {
        const __bf16* A = pass ? A2 : A1;
        const __bf16* W = pass ? W2 : W1;
#pragma unroll 4
        for (int k0 = 0; k0 < CC; k0 += 32) {
            v16bf a0 = load_a_bf16(A, CC, m0,      k0);
            v16bf a1 = load_a_bf16(A, CC, m0 + 16, k0);
            v16bf b0 = load_b_nk_bf16(W, CC, n0 +  0, k0);
            v16bf b1v = load_b_nk_bf16(W, CC, n0 + 16, k0);
            v16bf b2v = load_b_nk_bf16(W, CC, n0 + 32, k0);
            v16bf b3v = load_b_nk_bf16(W, CC, n0 + 48, k0);
            c00 = wmma_bf16(a0, b0, c00);
            c10 = wmma_bf16(a1, b0, c10);
            c01 = wmma_bf16(a0, b1v, c01);
            c11 = wmma_bf16(a1, b1v, c11);
            c02 = wmma_bf16(a0, b2v, c02);
            c12 = wmma_bf16(a1, b2v, c12);
            c03 = wmma_bf16(a0, b3v, c03);
            c13 = wmma_bf16(a1, b3v, c13);
        }
    }

    int l = lane_id();
    int nl = l & 15;
    int moff = (l >> 4) * 8;
    v8f acc[2][4] = {{c00, c01, c02, c03}, {c10, c11, c12, c13}};
#pragma unroll
    for (int r = 0; r < 2; ++r) {
#pragma unroll
        for (int j = 0; j < 4; ++j) {
            int n = n0 + j * 16 + nl;
            float bv = b1[n] + b2[n];
#pragma unroll
            for (int vv = 0; vv < 8; ++vv) {
                int m = m0 + r * 16 + vv + moff;
                out[(size_t)m * CC + n] = acc[r][j][vv] + bv;
            }
        }
    }
}

// =====================================================================
extern "C" void kernel_launch(void* const* d_in, const int* in_sizes, int n_in,
                              void* d_out, int out_size, void* d_ws, size_t ws_size,
                              hipStream_t stream) {
    (void)in_sizes; (void)n_in; (void)out_size; (void)ws_size;

    const float* x  = (const float*)d_in[0];
    const float* d  = (const float*)d_in[1];
    const float* Wf[8]; const float* bf[8];
    for (int i = 0; i < 8; ++i) {
        Wf[i] = (const float*)d_in[2 + 2 * i];   // k1,q1,v1,p1,k2,q2,v2,p2
        bf[i] = (const float*)d_in[3 + 2 * i];
    }

    // -------- workspace layout (bf16 elements) --------
    const size_t SZ  = (size_t)MM * CC;   // 4 Mi elems per activation tensor
    const size_t WSZ = (size_t)CC * CC;   // 256 Ki elems per weight
    __bf16* ws  = (__bf16*)d_ws;
    __bf16* xb  = ws + 0 * SZ;
    __bf16* db  = ws + 1 * SZ;
    __bf16* k1  = ws + 2 * SZ;
    __bf16* q1  = ws + 3 * SZ;
    __bf16* v1t = ws + 4 * SZ;
    __bf16* k2  = ws + 5 * SZ;
    __bf16* q2  = ws + 6 * SZ;
    __bf16* v2t = ws + 7 * SZ;
    __bf16* a1  = ws + 8 * SZ;            // attn(q2,k1,v1) in [B,T,C]
    __bf16* a2  = ws + 9 * SZ;            // attn(q1,k2,v2) in [B,T,C]
    __bf16* Wb[8];
    for (int i = 0; i < 8; ++i) Wb[i] = ws + 10 * SZ + (size_t)i * WSZ;

    dim3 blk(256);

    // -------- precision prep: one-time f32 -> bf16 --------
    cvt_bf16_kernel<<<(int)(SZ / 1024), blk, 0, stream>>>(x, xb, (int)SZ);
    cvt_bf16_kernel<<<(int)(SZ / 1024), blk, 0, stream>>>(d, db, (int)SZ);
    for (int i = 0; i < 8; ++i)
        cvt_bf16_kernel<<<(int)(WSZ / 1024), blk, 0, stream>>>(Wf[i], Wb[i], (int)WSZ);

    // -------- projections (k/q head-major, v transposed) --------
    dim3 grd(256);   // (M/32)*(C/64) wave-tiles / 8 waves
    proj_kernel<<<grd, blk, 0, stream>>>(xb, Wb[0], bf[0], k1,  0);
    proj_kernel<<<grd, blk, 0, stream>>>(xb, Wb[1], bf[1], q1,  0);
    proj_kernel<<<grd, blk, 0, stream>>>(xb, Wb[2], bf[2], v1t, 1);
    proj_kernel<<<grd, blk, 0, stream>>>(db, Wb[4], bf[4], k2,  0);
    proj_kernel<<<grd, blk, 0, stream>>>(db, Wb[5], bf[5], q2,  0);
    proj_kernel<<<grd, blk, 0, stream>>>(db, Wb[6], bf[6], v2t, 1);

    // -------- flash attention, both branches --------
    dim3 agrd(BB * HH * (TT / 128));  // 512
    attn_kernel<<<agrd, blk, 0, stream>>>(q2, k1, v1t, a1);  // branch 1
    attn_kernel<<<agrd, blk, 0, stream>>>(q1, k2, v2t, a2);  // branch 2

    // -------- fused output projections --------
    out_proj_kernel<<<grd, blk, 0, stream>>>(a1, Wb[3], bf[3], a2, Wb[7], bf[7],
                                             (float*)d_out);
}
